// MiniBatchGraphSAGE_59734405152779
// MI455X (gfx1250) — matrix operand order repaired and checked by reading, not compiled, not run on hardware
//
#include <hip/hip_runtime.h>

// ---------------------------------------------------------------------------
// GraphSAGE (3x SAGEConv) for MI455X / gfx1250, wave32.
// Scatter-bound workload: per-layer E*512B gather + 102M f32 atomics (L2
// resident). GEMMs (8.2 GFLOP total) fused with mean/bias/ReLU epilogue and
// executed with v_wmma_f32_16x16x32_bf16 (fp32 accumulate).
// ---------------------------------------------------------------------------

#define N_NODES 50000
#define E_EDGES 800000
#define D_FEAT  128   // input/hidden width (K per source matrix)

typedef __attribute__((ext_vector_type(16))) __bf16 v16bf;
typedef __attribute__((ext_vector_type(8)))  float  v8f;

union FragBF {
  v16bf v;
  __bf16 e[16];
  uint4  q[2];
};

// -------------------------------- utility kernels --------------------------

__global__ void zero_f4(float4* __restrict__ p, int n4) {
  int i = blockIdx.x * blockDim.x + threadIdx.x;
  if (i < n4) p[i] = float4{0.f, 0.f, 0.f, 0.f};
}

__global__ void deg_kernel(const int* __restrict__ dst, float* __restrict__ deg, int E) {
  int e = blockIdx.x * blockDim.x + threadIdx.x;
  if (e < E) atomicAdd(&deg[dst[e]], 1.0f);
}

__global__ void rdeg_kernel(float* __restrict__ deg, int n) {
  int i = blockIdx.x * blockDim.x + threadIdx.x;
  if (i < n) deg[i] = 1.0f / fmaxf(deg[i], 1.0f);
}

// Build BT[n][k] (bf16, k=0..255): k<128 -> Wl[k][n], else Wr[k-128][n].
__global__ void bt_kernel(const float* __restrict__ Wl, const float* __restrict__ Wr,
                          __bf16* __restrict__ BT, int dout) {
  int idx = blockIdx.x * blockDim.x + threadIdx.x;
  if (idx >= dout * 256) return;
  int n = idx >> 8, k = idx & 255;
  float w = (k < D_FEAT) ? Wl[k * dout + n] : Wr[(k - D_FEAT) * dout + n];
  BT[idx] = (__bf16)w;
}

// One wave per edge: lane l loads float4 l of the 128-float source row
// (512B coalesced per wave), then 4 global_atomic_add_f32 into agg[dst].
__global__ __launch_bounds__(256) void scatter_kernel(
    const float* __restrict__ X, const int* __restrict__ src,
    const int* __restrict__ dst, float* __restrict__ agg, int E) {
  int e = blockIdx.x * 8 + (threadIdx.x >> 5);
  if (e >= E) return;
  int lane = threadIdx.x & 31;
  int s = src[e], d = dst[e];
  float4 v = ((const float4*)(X + (size_t)s * D_FEAT))[lane];
  float* a = agg + (size_t)d * D_FEAT + lane * 4;
  atomicAdd(a + 0, v.x);
  atomicAdd(a + 1, v.y);
  atomicAdd(a + 2, v.z);
  atomicAdd(a + 3, v.w);
}

// ------------------------------ fused WMMA GEMM -----------------------------
// Y[m][n] = act( sum_k [agg[m]*rdeg[m] | X[m]][k] * BT[n][k] + bias[n] )
// 4 waves/block, wave owns a 16-row strip; K = 256 in 8 bf16 WMMA steps.
template <int DOUT, bool RELU>
__global__ __launch_bounds__(128) void sage_gemm(
    const float* __restrict__ agg, const float* __restrict__ rdeg,
    const float* __restrict__ X, const __bf16* __restrict__ BT,
    const float* __restrict__ bias, float* __restrict__ Y, int Nn) {
  const int lane = threadIdx.x & 31;
  const int wave = threadIdx.x >> 5;
  const int lh   = lane >> 4;                       // lane half
  const int m0   = (blockIdx.x * 4 + wave) * 16;

  int ra = m0 + (lane & 15);
  if (ra > Nn - 1) ra = Nn - 1;                     // clamp; stores are guarded
  const float scale = rdeg[ra];

  // A fragments (ISA 16-bit A layout: per lane K chunks at lh*8 and 16+lh*8).
  FragBF afrag[8];
#pragma unroll
  for (int kt = 0; kt < 8; ++kt) {
    const float* srcp;
    float sc;
    int c0;
    if (kt < 4) { srcp = agg + (size_t)ra * D_FEAT; sc = scale; c0 = kt * 32 + lh * 8; }
    else        { srcp = X   + (size_t)ra * D_FEAT; sc = 1.0f;  c0 = (kt - 4) * 32 + lh * 8; }
    float4 f0 = ((const float4*)(srcp + c0))[0];
    float4 f1 = ((const float4*)(srcp + c0))[1];
    float4 f2 = ((const float4*)(srcp + c0 + 16))[0];
    float4 f3 = ((const float4*)(srcp + c0 + 16))[1];
    float t[16] = {f0.x, f0.y, f0.z, f0.w, f1.x, f1.y, f1.z, f1.w,
                   f2.x, f2.y, f2.z, f2.w, f3.x, f3.y, f3.z, f3.w};
#pragma unroll
    for (int i = 0; i < 16; ++i) afrag[kt].e[i] = (__bf16)(t[i] * sc);
  }

  const int ncol = (lane & 15);
#pragma unroll
  for (int nt = 0; nt < DOUT / 16; ++nt) {
    v8f acc = {0.f, 0.f, 0.f, 0.f, 0.f, 0.f, 0.f, 0.f};
    const int n = nt * 16 + ncol;
#pragma unroll
    for (int kt = 0; kt < 8; ++kt) {
      // B fragment (ISA 16-bit B layout): K chunks at kt*32 + lh*8 and kt*32 + 16 + lh*8.
      FragBF bfrag;
      const uint4* bp = (const uint4*)(BT + (size_t)n * 256 + kt * 32 + lh * 8);
      bfrag.q[0] = bp[0];
      bfrag.q[1] = bp[2];
      acc = __builtin_amdgcn_wmma_f32_16x16x32_bf16(
          /*neg_a=*/false, afrag[kt].v, /*neg_b=*/false, bfrag.v,
          /*c_mod=*/(short)0, acc, /*reuse_a=*/false, /*reuse_b=*/false);
    }
    const float bcol = bias[n];
#pragma unroll
    for (int i = 0; i < 8; ++i) {
      int row = m0 + i + lh * 8;                    // D layout: VGPR i -> M = i + half*8
      float val = acc[i] + bcol;
      if (RELU) val = fmaxf(val, 0.0f);
      if (row < Nn) Y[(size_t)row * DOUT + n] = val;
    }
  }
}

// --------------------------------- launcher --------------------------------

extern "C" void kernel_launch(void* const* d_in, const int* in_sizes, int n_in,
                              void* d_out, int out_size, void* d_ws, size_t ws_size,
                              hipStream_t stream) {
  const float* x   = (const float*)d_in[0];
  const int*   ei  = (const int*)d_in[1];
  const float* Wl0 = (const float*)d_in[2];
  const float* Wr0 = (const float*)d_in[3];
  const float* b0  = (const float*)d_in[4];
  const float* Wl1 = (const float*)d_in[5];
  const float* Wr1 = (const float*)d_in[6];
  const float* b1  = (const float*)d_in[7];
  const float* Wl2 = (const float*)d_in[8];
  const float* Wr2 = (const float*)d_in[9];
  const float* b2  = (const float*)d_in[10];

  const int* src = ei;             // edge_index[0]
  const int* dst = ei + E_EDGES;   // edge_index[1]

  // Workspace carve (all 16B aligned): deg | agg | h0 | h1 | BT0 | BT1 | BT2
  float*  deg = (float*)d_ws;
  float*  agg = deg + N_NODES;
  float*  h0  = agg + (size_t)N_NODES * D_FEAT;
  float*  h1  = h0  + (size_t)N_NODES * D_FEAT;
  __bf16* BT0 = (__bf16*)(h1 + (size_t)N_NODES * D_FEAT);
  __bf16* BT1 = BT0 + 128 * 256;
  __bf16* BT2 = BT1 + 128 * 256;

  const int DEG4 = N_NODES / 4;                  // 12500
  const int AGG4 = N_NODES * D_FEAT / 4;         // 1,600,000
  const int scat_grid = (E_EDGES + 7) / 8;
  const int gemm_grid = (N_NODES + 63) / 64;

  // Degrees (shared by all layers), then reciprocal in place.
  zero_f4<<<(DEG4 + 255) / 256, 256, 0, stream>>>((float4*)deg, DEG4);
  deg_kernel<<<(E_EDGES + 255) / 256, 256, 0, stream>>>(dst, deg, E_EDGES);
  rdeg_kernel<<<(N_NODES + 255) / 256, 256, 0, stream>>>(deg, N_NODES);

  // Weight transposes -> bf16 BT buffers.
  bt_kernel<<<(128 * 256 + 255) / 256, 256, 0, stream>>>(Wl0, Wr0, BT0, 128);
  bt_kernel<<<(128 * 256 + 255) / 256, 256, 0, stream>>>(Wl1, Wr1, BT1, 128);
  bt_kernel<<<(64 * 256 + 255) / 256, 256, 0, stream>>>(Wl2, Wr2, BT2, 64);

  // Layer 0: x -> h0
  zero_f4<<<(AGG4 + 255) / 256, 256, 0, stream>>>((float4*)agg, AGG4);
  scatter_kernel<<<scat_grid, 256, 0, stream>>>(x, src, dst, agg, E_EDGES);
  sage_gemm<128, true><<<gemm_grid, 128, 0, stream>>>(agg, deg, x, BT0, b0, h0, N_NODES);

  // Layer 1: h0 -> h1
  zero_f4<<<(AGG4 + 255) / 256, 256, 0, stream>>>((float4*)agg, AGG4);
  scatter_kernel<<<scat_grid, 256, 0, stream>>>(h0, src, dst, agg, E_EDGES);
  sage_gemm<128, true><<<gemm_grid, 128, 0, stream>>>(agg, deg, h0, BT1, b1, h1, N_NODES);

  // Layer 2: h1 -> out (no ReLU)
  zero_f4<<<(AGG4 + 255) / 256, 256, 0, stream>>>((float4*)agg, AGG4);
  scatter_kernel<<<scat_grid, 256, 0, stream>>>(h1, src, dst, agg, E_EDGES);
  sage_gemm<64, false><<<gemm_grid, 128, 0, stream>>>(agg, deg, h1, BT2, b2,
                                                      (float*)d_out, N_NODES);
}
